// Triangle_Att_Start_34660386079039
// MI455X (gfx1250) — matrix operand order repaired and checked by reading, not compiled, not run on hardware
//
#include <hip/hip_runtime.h>
#include <hip/hip_bf16.h>

typedef __bf16 bf16;
typedef __attribute__((ext_vector_type(16))) __bf16 v16bf;
typedef __attribute__((ext_vector_type(8)))  __bf16 v8bf;
typedef __attribute__((ext_vector_type(8)))  float  v8f;

#define EPSLN 1e-5f

static __device__ __forceinline__ v16bf cat8(v8bf lo, v8bf hi) {
  return __builtin_shufflevector(lo, hi, 0,1,2,3,4,5,6,7,8,9,10,11,12,13,14,15);
}
static __device__ __forceinline__ v8f wmma_bf16(v16bf a, v16bf b, v8f c) {
  // D = A(16x32 bf16) * B(32x16 bf16) + C(16x16 f32)
  return __builtin_amdgcn_wmma_f32_16x16x32_bf16(false, a, false, b, (short)0, c, false, false);
}
// A fragment: row-contiguous source, K-chunk base kc (multiple of 32 halves).
// lane<16: halves K = kc+0..15; lane>=16: K = kc+16..31
static __device__ __forceinline__ v16bf load_afrag(const bf16* rowBase, int kc, int hs) {
  const bf16* p = rowBase + kc + (hs << 4);
  return cat8(*(const v8bf*)p, *(const v8bf*)(p + 8));
}
// B fragment: column-contiguous source (col stored K-major). lane = column.
// lane<16: K = kc+0..15 ; lane>=16: K = kc+16..31
static __device__ __forceinline__ v16bf load_bfrag(const bf16* colBase, int kc, int hs) {
  const bf16* p = colBase + kc + (hs << 4);
  return cat8(*(const v8bf*)p, *(const v8bf*)(p + 8));
}
static __device__ __forceinline__ float fast_rcp(float x) {
  return __builtin_amdgcn_rcpf(x);
}

// ---------------- TDM: DMA a contiguous 2-byte-element tile into LDS ----------------
#if __has_builtin(__builtin_amdgcn_tensor_load_to_lds) && __has_builtin(__builtin_amdgcn_s_wait_tensorcnt)
#define USE_TDM 1
typedef unsigned int v4ui __attribute__((ext_vector_type(4)));
typedef int v4i __attribute__((ext_vector_type(4)));
typedef int v8i __attribute__((ext_vector_type(8)));

static __device__ __forceinline__ void tdm_load_1d(unsigned int lds_off,
                                                   const void* gaddr,
                                                   unsigned int nelem) {
  unsigned long long ga = (unsigned long long)(uintptr_t)gaddr;
  // D# group 0: count=1 (valid user descriptor), lds_addr, 57-bit global_addr, type=2
  v4ui g0;
  g0[0] = 1u;
  g0[1] = lds_off;
  g0[2] = (unsigned int)(ga & 0xffffffffu);
  g0[3] = (unsigned int)((ga >> 32) & 0x01ffffffu) | (2u << 30);
  // D# group 1: workgroup_mask=0, data_size=1 (2B), tensor_dim0=nelem,
  // tensor_dim1=1, tile_dim0=nelem (<=65535), tile_dim1=1, dim0_stride=nelem
  v8i g1;
  g1[0] = (int)(1u << 16);                                    // data_size = 2 bytes
  g1[1] = (int)((nelem & 0xffffu) << 16);                     // tensor_dim0[15:0]
  g1[2] = (int)(((nelem >> 16) & 0xffffu) | (1u << 16));      // dim0[31:16] | tensor_dim1=1
  g1[3] = (int)((nelem & 0xffffu) << 16);                     // tile_dim0
  g1[4] = 1;                                                  // tile_dim1 = 1
  g1[5] = (int)nelem;                                         // tensor_dim0_stride lo
  g1[6] = 0;
  g1[7] = 0;
  v4i z4 = {0, 0, 0, 0};
#if __clang_major__ >= 23
  v8i z8 = {0, 0, 0, 0, 0, 0, 0, 0};
  __builtin_amdgcn_tensor_load_to_lds(g0, g1, z4, z4, z8, 0);
#else
  __builtin_amdgcn_tensor_load_to_lds(g0, g1, z4, z4, 0);
#endif
}
#else
#define USE_TDM 0
#endif

// ---------------- Phase 0: weight transpose/convert ----------------
// Wt[c][k] (512x128 bf16): c in [0,512): 0..127 Wq, 128..255 Wk, 256..383 Wv, 384..511 Wg
// Wot[n][k] (128x128 bf16)
__global__ void k_prep(const float* __restrict__ Wq, const float* __restrict__ Wk,
                       const float* __restrict__ Wv, const float* __restrict__ Wg,
                       const float* __restrict__ Wo,
                       bf16* __restrict__ Wt, bf16* __restrict__ Wot) {
  int idx = blockIdx.x * blockDim.x + threadIdx.x;
  if (idx < 512 * 128) {
    int c = idx >> 7, k = idx & 127;
    const float* W = (c < 128) ? Wq : (c < 256) ? Wk : (c < 384) ? Wv : Wg;
    Wt[idx] = (bf16)W[k * 128 + (c & 127)];
  } else if (idx < 512 * 128 + 128 * 128) {
    int t = idx - 512 * 128;
    int nn = t >> 7, k = t & 127;
    Wot[t] = (bf16)Wo[k * 128 + nn];
  }
}

// ---------------- Phase A: LayerNorm + QKVG/bias projections ----------------
// grid: 512 blocks x 256 threads (8 waves x 16 rows = 128 positions/block)
// dynamic LDS: 8*16*128 bf16 = 32KB (launch passes 64KB)
__global__ __launch_bounds__(256)
void k_proj(const float* __restrict__ pair, const float* __restrict__ ln_g,
            const float* __restrict__ ln_b, const float* __restrict__ Wb,
            const float* __restrict__ bg, const bf16* __restrict__ Wt,
            bf16* __restrict__ Qh, bf16* __restrict__ Kh,
            bf16* __restrict__ Vt, bf16* __restrict__ Gh,
            float* __restrict__ Bt) {
  extern __shared__ char smem_raw[];
  bf16* xs = (bf16*)smem_raw;
  const int tid = threadIdx.x, lane = tid & 31, w = tid >> 5;
  const int hs = lane >> 4, ln = lane & 15;
  const int p0 = blockIdx.x * 128 + w * 16;
  bf16* xw = xs + w * 16 * 128;

  // --- LayerNorm 16 rows, whole wave per row ---
  for (int r = 0; r < 16; ++r) {
    const float4 v = *(const float4*)(pair + (size_t)(p0 + r) * 128 + lane * 4);
    float s  = v.x + v.y + v.z + v.w;
    float ss = v.x * v.x + v.y * v.y + v.z * v.z + v.w * v.w;
    for (int off = 16; off; off >>= 1) {
      s  += __shfl_xor(s,  off, 32);
      ss += __shfl_xor(ss, off, 32);
    }
    float mu   = s * (1.0f / 128.0f);
    float var  = ss * (1.0f / 128.0f) - mu * mu;
    float rstd = rsqrtf(var + EPSLN);
    int c = lane * 4;
    xw[r * 128 + c + 0] = (bf16)(((v.x - mu) * rstd) * ln_g[c + 0] + ln_b[c + 0]);
    xw[r * 128 + c + 1] = (bf16)(((v.y - mu) * rstd) * ln_g[c + 1] + ln_b[c + 1]);
    xw[r * 128 + c + 2] = (bf16)(((v.z - mu) * rstd) * ln_g[c + 2] + ln_b[c + 2]);
    xw[r * 128 + c + 3] = (bf16)(((v.w - mu) * rstd) * ln_g[c + 3] + ln_b[c + 3]);
  }
  __syncthreads();

  // --- bias projection x @ Wb -> Bt[n][p2][p1] (VALU, tiny) ---
  for (int r = 0; r < 16; ++r) {
    float acc[4] = {0.f, 0.f, 0.f, 0.f};
    int k = lane * 4;
    #pragma unroll
    for (int q = 0; q < 4; ++q) {
      float xk = (float)xw[r * 128 + k + q];
      #pragma unroll
      for (int nn = 0; nn < 4; ++nn) acc[nn] += xk * Wb[(k + q) * 4 + nn];
    }
    #pragma unroll
    for (int nn = 0; nn < 4; ++nn)
      for (int off = 16; off; off >>= 1) acc[nn] += __shfl_xor(acc[nn], off, 32);
    if (lane == 0) {
      int gp = p0 + r, p1 = gp >> 8, p2 = gp & 255;
      #pragma unroll
      for (int nn = 0; nn < 4; ++nn) Bt[((size_t)nn * 256 + p2) * 256 + p1] = acc[nn];
    }
  }

  // --- main GEMM: x(16x128) @ Wt^T(128x512) with WMMA ---
  v16bf afr[4];
  #pragma unroll
  for (int c = 0; c < 4; ++c) afr[c] = load_afrag(xw + ln * 128, c * 32, hs);

  for (int nt = 0; nt < 32; ++nt) {
    v8bf blo[4], bhi[4];
    const bf16* colp = Wt + (size_t)(nt * 16 + ln) * 128 + (hs << 4);
    #pragma unroll
    for (int c = 0; c < 4; ++c) {
      blo[c] = *(const v8bf*)(colp + c * 32);
      bhi[c] = *(const v8bf*)(colp + c * 32 + 8);
    }
    v8f acc = {};
    #pragma unroll
    for (int c = 0; c < 4; ++c) acc = wmma_bf16(afr[c], cat8(blo[c], bhi[c]), acc);

    int col = nt * 16 + ln;
    int type = nt >> 3;
    int cc = col & 127;
    int hn = cc >> 5, hh = cc & 31;
    #pragma unroll
    for (int r = 0; r < 8; ++r) {
      int gp = p0 + r + (hs << 3);
      int p1 = gp >> 8, p2 = gp & 255;
      float vfl = acc[r];
      if (type == 0) {
        Qh[(((size_t)p1 * 4 + hn) * 256 + p2) * 32 + hh] = (bf16)vfl;
      } else if (type == 1) {
        Kh[(((size_t)p1 * 4 + hn) * 256 + p2) * 32 + hh] = (bf16)vfl;
      } else if (type == 2) {
        Vt[(((size_t)p1 * 4 + hn) * 32 + hh) * 256 + p2] = (bf16)vfl;
      } else {
        float gv = vfl + bg[cc];
        gv = fast_rcp(1.0f + __expf(-gv));
        Gh[(((size_t)p1 * 4 + hn) * 256 + p2) * 32 + hh] = (bf16)gv;
      }
    }
  }
}

// ---------------- Phase B: attention per (head, i) ----------------
// grid: 1024 blocks x 256 threads
// dynamic LDS: K(8192) + V(8192) + P(8*4096) halves = 49152 halves = 96KB
// K tile at LDS offset 0, V^T tile at LDS offset 16384 bytes (TDM descriptors).
__global__ __launch_bounds__(256)
void k_attn(const bf16* __restrict__ Qh, const bf16* __restrict__ Kh,
            const bf16* __restrict__ Vt, const bf16* __restrict__ Gh,
            const float* __restrict__ Bt, bf16* __restrict__ Og) {
  extern __shared__ char smem_raw[];
  bf16* Ks = (bf16*)smem_raw;            // 256 keys x 32 h   (K-major rows)
  bf16* Vs = Ks + 8192;                  // 32 h x 256 keys   (transposed V)
  bf16* Ps = Vs + 8192;                  // 8 waves x 16 x 256 probs
  const int tid = threadIdx.x, lane = tid & 31, w = tid >> 5;
  const int hs = lane >> 4, ln = lane & 15;
  const int n = blockIdx.x & 3, i = blockIdx.x >> 2;
  const bf16* Qb = Qh + ((size_t)i * 4 + n) * 256 * 32;
  const bf16* Kb = Kh + ((size_t)i * 4 + n) * 256 * 32;
  const bf16* Vb = Vt + ((size_t)i * 4 + n) * 32 * 256;
  const bf16* Gb = Gh + ((size_t)i * 4 + n) * 256 * 32;
  const float* Bn = Bt + (size_t)n * 256 * 256;
  bf16* Pw = Ps + w * 16 * 256;
  const float factor = 0.17677669529663687f; // 1/sqrt(32)

#if USE_TDM
  // Tensor Data Mover: DMA the 16KB K and V^T tiles into LDS (wave 0 issues,
  // TENSORcnt tracks completion, barrier publishes to the workgroup).
  if (w == 0) {
    tdm_load_1d(0u,     Kb, 8192u);
    tdm_load_1d(16384u, Vb, 8192u);
    __builtin_amdgcn_s_wait_tensorcnt(0);
  }
#else
  #pragma unroll
  for (int t = 0; t < 4; ++t) {
    int c = tid + t * 256;
    *(v8bf*)(Ks + c * 8) = *(const v8bf*)(Kb + c * 8);
    *(v8bf*)(Vs + c * 8) = *(const v8bf*)(Vb + c * 8);
  }
#endif
  __syncthreads();

  for (int s = 0; s < 2; ++s) {
    int j0 = w * 32 + s * 16;
    // Q A-fragment: full K=32
    v16bf aq = load_afrag(Qb + (size_t)(j0 + ln) * 32, 0, hs);

    // S = Q K^T * factor + bias  (one WMMA per 16x16 tile)
    float S[16][8];
    for (int kt = 0; kt < 16; ++kt) {
      v16bf bk = load_bfrag(Ks + (kt * 16 + ln) * 32, 0, hs);
      v8f z = {};
      v8f sacc = wmma_bf16(aq, bk, z);
      #pragma unroll
      for (int r = 0; r < 8; ++r) {
        int j = j0 + r + (hs << 3);
        int kk = kt * 16 + ln;
        S[kt][r] = sacc[r] * factor + Bn[(size_t)j * 256 + kk];
      }
    }
    // softmax over k: row lives across the 16-lane half at fixed VGPR r
    #pragma unroll
    for (int r = 0; r < 8; ++r) {
      float mm = -3.0e38f;
      for (int kt = 0; kt < 16; ++kt) mm = fmaxf(mm, S[kt][r]);
      for (int off = 1; off < 16; off <<= 1) mm = fmaxf(mm, __shfl_xor(mm, off, 32));
      float sum = 0.f;
      for (int kt = 0; kt < 16; ++kt) { float e = __expf(S[kt][r] - mm); S[kt][r] = e; sum += e; }
      for (int off = 1; off < 16; off <<= 1) sum += __shfl_xor(sum, off, 32);
      float inv = fast_rcp(sum);
      for (int kt = 0; kt < 16; ++kt) S[kt][r] *= inv;
    }
    // write P (bf16) to LDS in row-major [j-in-strip][k]
    for (int kt = 0; kt < 16; ++kt) {
      int kk = kt * 16 + ln;
      #pragma unroll
      for (int r = 0; r < 8; ++r) {
        int M = r + (hs << 3);
        Pw[M * 256 + kk] = (bf16)S[kt][r];
      }
    }
    __syncthreads();
    // O = P @ V, then gate, store bf16
    for (int ht = 0; ht < 2; ++ht) {
      v8f acc = {};
      #pragma unroll
      for (int c = 0; c < 8; ++c) {
        v16bf ap = load_afrag(Pw + ln * 256, c * 32, hs);
        v16bf bv = load_bfrag(Vs + (ht * 16 + ln) * 256, c * 32, hs);
        acc = wmma_bf16(ap, bv, acc);
      }
      #pragma unroll
      for (int r = 0; r < 8; ++r) {
        int j = j0 + r + (hs << 3);
        int h = ht * 16 + ln;
        float g = (float)Gb[(size_t)j * 32 + h];
        Og[((size_t)i * 256 + j) * 128 + n * 32 + h] = (bf16)(acc[r] * g);
      }
    }
    __syncthreads();
  }
}

// ---------------- Phase C: output projection ----------------
// grid: 512 blocks x 256 threads
__global__ __launch_bounds__(256)
void k_outproj(const bf16* __restrict__ Og, const bf16* __restrict__ Wot,
               const float* __restrict__ bo, float* __restrict__ out) {
  const int tid = threadIdx.x, lane = tid & 31, w = tid >> 5;
  const int hs = lane >> 4, ln = lane & 15;
  const int p0 = blockIdx.x * 128 + w * 16;
  v16bf afr[4];
  #pragma unroll
  for (int c = 0; c < 4; ++c) afr[c] = load_afrag(Og + (size_t)(p0 + ln) * 128, c * 32, hs);
  for (int nt = 0; nt < 8; ++nt) {
    v8bf blo[4], bhi[4];
    const bf16* colp = Wot + (size_t)(nt * 16 + ln) * 128 + (hs << 4);
    #pragma unroll
    for (int c = 0; c < 4; ++c) {
      blo[c] = *(const v8bf*)(colp + c * 32);
      bhi[c] = *(const v8bf*)(colp + c * 32 + 8);
    }
    v8f acc = {};
    #pragma unroll
    for (int c = 0; c < 4; ++c) acc = wmma_bf16(afr[c], cat8(blo[c], bhi[c]), acc);

    int col = nt * 16 + ln;
    float bias = bo[col];
    #pragma unroll
    for (int r = 0; r < 8; ++r) {
      int row = p0 + r + (hs << 3);
      out[(size_t)row * 128 + col] = acc[r] + bias;
    }
  }
}

extern "C" void kernel_launch(void* const* d_in, const int* in_sizes, int n_in,
                              void* d_out, int out_size, void* d_ws, size_t ws_size,
                              hipStream_t stream) {
  (void)in_sizes; (void)n_in; (void)out_size; (void)ws_size;
  const float* pair = (const float*)d_in[0];
  const float* ln_g = (const float*)d_in[1];
  const float* ln_b = (const float*)d_in[2];
  const float* Wq   = (const float*)d_in[3];
  const float* Wk   = (const float*)d_in[4];
  const float* Wv   = (const float*)d_in[5];
  const float* Wb   = (const float*)d_in[6];
  const float* Wg   = (const float*)d_in[7];
  const float* bg   = (const float*)d_in[8];
  const float* Wo   = (const float*)d_in[9];
  const float* bo   = (const float*)d_in[10];
  float* out = (float*)d_out;

  char* ws = (char*)d_ws;
  size_t off = 0;
  bf16*  Wt  = (bf16*)(ws + off); off += (size_t)512 * 128 * 2;          // 128 KB
  bf16*  Wot = (bf16*)(ws + off); off += (size_t)128 * 128 * 2;          // 32 KB
  bf16*  Qh  = (bf16*)(ws + off); off += (size_t)256 * 4 * 256 * 32 * 2; // 16 MB
  bf16*  Kh  = (bf16*)(ws + off); off += (size_t)256 * 4 * 256 * 32 * 2;
  bf16*  Vtr = (bf16*)(ws + off); off += (size_t)256 * 4 * 32 * 256 * 2;
  bf16*  Gh  = (bf16*)(ws + off); off += (size_t)256 * 4 * 256 * 32 * 2;
  float* Bt  = (float*)(ws + off); off += (size_t)4 * 256 * 256 * 4;     // 1 MB
  bf16*  Og  = (bf16*)(ws + off); off += (size_t)65536 * 128 * 2;        // 16 MB

  k_prep<<<(512 * 128 + 128 * 128 + 255) / 256, 256, 0, stream>>>(Wq, Wk, Wv, Wg, Wo, Wt, Wot);
  k_proj<<<512, 256, 65536, stream>>>(pair, ln_g, ln_b, Wb, bg, Wt, Qh, Kh, Vtr, Gh, Bt);
  k_attn<<<1024, 256, 98304, stream>>>(Qh, Kh, Vtr, Gh, Bt, Og);
  k_outproj<<<512, 256, 0, stream>>>(Og, Wot, bo, out);
}
